// WordNet_26379689132255
// MI455X (gfx1250) — compile-verified, matrix operand order checked by reading, not run on hardware
//
#include <hip/hip_runtime.h>
#include <hip/hip_bf16.h>
#include <math.h>

// ---------------------------------------------------------------------------
// Problem constants (from reference)
// ---------------------------------------------------------------------------
#define NW_   50000
#define NS_   10000
#define E_    100000
#define HH    6
#define CC    300
#define HC    1800         // H*C
#define HCP   1824         // HC padded to multiple of 32 (K of final GEMM)
#define DS    640          // sentence feature dim
#define DW    300          // word feature dim / out dim
#define NP    304          // out dim padded to multiple of 16
#define DH    2048         // FFN hidden (folded away)

typedef __attribute__((ext_vector_type(8)))  __bf16 v8bf;
typedef __attribute__((ext_vector_type(16))) __bf16 v16bf;
typedef __attribute__((ext_vector_type(8)))  float  v8f;

struct BfPair { v8bf lo, hi; };

__device__ __forceinline__ int imin(int a, int b) { return a < b ? a : b; }

__device__ __forceinline__ unsigned short f2bf(float f) {
    unsigned int u = __float_as_uint(f);
    unsigned int r = u + 0x7FFFu + ((u >> 16) & 1u);   // round-to-nearest-even
    return (unsigned short)(r >> 16);
}

// monotone float<->uint map for atomicMax-based segment max
__device__ __forceinline__ unsigned int enc_f32(float f) {
    unsigned int u = __float_as_uint(f);
    return (u & 0x80000000u) ? ~u : (u | 0x80000000u);
}
__device__ __forceinline__ float dec_f32(unsigned int e) {
    unsigned int u = (e & 0x80000000u) ? (e & 0x7FFFFFFFu) : ~e;
    return __uint_as_float(u);
}

__device__ __forceinline__ float wave_sum(float v) {
    for (int off = 16; off > 0; off >>= 1) v += __shfl_down(v, off, 32);
    return v;
}

// ---------------------------------------------------------------------------
// bf16 WMMA fragment load: 16x32 A tile (or 32x16 B tile via its row-major
// [N,K] weight layout). Per ISA 7.12.2: lane L holds row (L&15), K-halves
// {kb..kb+7} and {kb+16..kb+23} with kb = (L>>4)*8 -> two 16B loads.
// ---------------------------------------------------------------------------
__device__ __forceinline__ v16bf load_frag_bf16(const unsigned short* __restrict__ base,
                                                int ld, int row, int k) {
    const unsigned short* p = base + (size_t)row * ld + k;
    BfPair f;
    f.lo = *(const v8bf*)(p);
    f.hi = *(const v8bf*)(p + 16);
    return __builtin_bit_cast(v16bf, f);
}

__device__ __forceinline__ void store_acc(v8f acc, float* __restrict__ C, int ldc,
                                          int m0, int n0, int M, int Nvalid, int lane,
                                          const float* __restrict__ bias,
                                          const float* __restrict__ resid, int ldr) {
    // C/D layout: col = lane&15, rows (lane>>4)*8 + i
    int col = n0 + (lane & 15);
    int r0  = m0 + ((lane >> 4) * 8);
    if (col >= Nvalid || m0 >= M) return;   // M is a multiple of 16 -> whole 16-row block valid or not
    float badd = bias ? bias[col] : 0.0f;
#pragma unroll
    for (int i = 0; i < 8; ++i) {
        int row = r0 + i;
        float v = acc[i] + badd;
        if (resid) v += resid[(size_t)row * ldr + col];
        C[(size_t)row * ldc + col] = v;
    }
}

// ---------------------------------------------------------------------------
// Generic bf16 GEMM: C[M,Nvalid] = A[M,K] * B[Npad,K]^T (+bias +resid)
// 128 threads = 4 waves; each wave computes a 32x32 tile (2x2 WMMA frags).
// K multiple of 32; B zero-padded to Npad (multiple of 16); row loads clamped.
// ---------------------------------------------------------------------------
__global__ __launch_bounds__(128) void k_wmma_gemm(
    const unsigned short* __restrict__ A, int lda, int M,
    const unsigned short* __restrict__ B, int ldb, int Npad,
    float* __restrict__ C, int ldc, int Nvalid, int K,
    const float* __restrict__ bias, const float* __restrict__ resid, int ldr)
{
    const int lane = threadIdx.x & 31;
    const int wave = threadIdx.x >> 5;
    const int m_base = blockIdx.y * 64 + (wave >> 1) * 32;
    const int n_base = blockIdx.x * 64 + (wave & 1) * 32;
    const int fr = lane & 15;
    const int kb = (lane >> 4) * 8;
    const int ra0 = imin(m_base + fr,      M - 1);
    const int ra1 = imin(m_base + 16 + fr, M - 1);
    const int rb0 = imin(n_base + fr,      Npad - 1);
    const int rb1 = imin(n_base + 16 + fr, Npad - 1);

    v8f acc00 = {}, acc01 = {}, acc10 = {}, acc11 = {};
    for (int kk = 0; kk < K; kk += 32) {
        v16bf a0 = load_frag_bf16(A, lda, ra0, kk + kb);
        v16bf a1 = load_frag_bf16(A, lda, ra1, kk + kb);
        v16bf b0 = load_frag_bf16(B, ldb, rb0, kk + kb);
        v16bf b1 = load_frag_bf16(B, ldb, rb1, kk + kb);
        acc00 = __builtin_amdgcn_wmma_f32_16x16x32_bf16(false, a0, false, b0, (short)0, acc00, false, false);
        acc01 = __builtin_amdgcn_wmma_f32_16x16x32_bf16(false, a0, false, b1, (short)0, acc01, false, false);
        acc10 = __builtin_amdgcn_wmma_f32_16x16x32_bf16(false, a1, false, b0, (short)0, acc10, false, false);
        acc11 = __builtin_amdgcn_wmma_f32_16x16x32_bf16(false, a1, false, b1, (short)0, acc11, false, false);
    }
    store_acc(acc00, C, ldc, m_base,      n_base,      M, Nvalid, lane, bias, resid, ldr);
    store_acc(acc01, C, ldc, m_base,      n_base + 16, M, Nvalid, lane, bias, resid, ldr);
    store_acc(acc10, C, ldc, m_base + 16, n_base,      M, Nvalid, lane, bias, resid, ldr);
    store_acc(acc11, C, ldc, m_base + 16, n_base + 16, M, Nvalid, lane, bias, resid, ldr);
}

// ---------------------------------------------------------------------------
// Small helper kernels
// ---------------------------------------------------------------------------
__global__ void k_zero_f32(float* __restrict__ p, long long n) {
    long long i = blockIdx.x * (long long)blockDim.x + threadIdx.x;
    if (i < n) p[i] = 0.0f;
}

// f32 -> bf16, zero-padding extra rows (dst has dstRows >= srcRows, same #cols)
__global__ void k_cvt_pad(const float* __restrict__ src, unsigned short* __restrict__ dst,
                          int srcRows, int cols, long long total) {
    long long idx = blockIdx.x * (long long)blockDim.x + threadIdx.x;
    if (idx >= total) return;
    long long r = idx / cols;
    float v = (r < srcRows) ? src[idx] : 0.0f;
    dst[idx] = f2bf(v);
}

// v[h,k] = sum_c att_dst[h,c] * Wdst[h*C+c, k]   (so a_d = Hw @ v^T, avoids xd)
__global__ void k_vdst(const float* __restrict__ Wdst, const float* __restrict__ att_dst,
                       float* __restrict__ v) {
    int idx = blockIdx.x * blockDim.x + threadIdx.x;
    if (idx >= HH * DW) return;
    int h = idx / DW, k = idx % DW;
    float s = 0.0f;
    for (int c = 0; c < CC; ++c)
        s += att_dst[h * CC + c] * Wdst[(size_t)(h * CC + c) * DW + k];
    v[idx] = s;
}

// Wf[o,i] = sum_j W2[o,j]*W1[j,i]  -> bf16, padded to [NP, HCP]
__global__ __launch_bounds__(256) void k_foldW(const float* __restrict__ W2,
                                               const float* __restrict__ W1,
                                               unsigned short* __restrict__ Wf) {
    int idx = blockIdx.x * blockDim.x + threadIdx.x;
    if (idx >= NP * HCP) return;
    int o = idx / HCP, i = idx % HCP;
    float s = 0.0f;
    if (o < DW && i < HC) {
        for (int j = 0; j < DH; ++j)
            s += W2[(size_t)o * DH + j] * W1[(size_t)j * HC + i];
    }
    Wf[idx] = f2bf(s);
}

// bf[o] = sum_j W2[o,j]*b1[j] + b2[o]
__global__ void k_foldb(const float* __restrict__ W2, const float* __restrict__ b1,
                        const float* __restrict__ b2, float* __restrict__ bf) {
    int o = blockIdx.x * blockDim.x + threadIdx.x;
    if (o >= NP) return;
    float s = 0.0f;
    if (o < DW) {
        for (int j = 0; j < DH; ++j) s += W2[(size_t)o * DH + j] * b1[j];
        s += b2[o];
    }
    bf[o] = s;
}

// a_s[n,h] = sum_c xs[n, h*C + c] * att_src[h,c]   (one wave per (n,h))
__global__ __launch_bounds__(256) void k_as(const float* __restrict__ xs,
                                            const float* __restrict__ att_src,
                                            float* __restrict__ a_s, int nRows) {
    int gw = blockIdx.x * 8 + (threadIdx.x >> 5);
    int lane = threadIdx.x & 31;
    if (gw >= nRows * HH) return;
    int n = gw / HH, h = gw % HH;
    const float* xrow = xs + (size_t)n * HCP + h * CC;
    const float* arow = att_src + h * CC;
    float acc = 0.0f;
    for (int c = lane; c < CC; c += 32) acc += xrow[c] * arow[c];
    acc = wave_sum(acc);
    if (lane == 0) a_s[gw] = acc;
}

// a_d[n,h] = sum_k Hw[n,k] * v[h,k]   (one wave per (n,h))
__global__ __launch_bounds__(256) void k_ad(const float* __restrict__ Hw,
                                            const float* __restrict__ v,
                                            float* __restrict__ a_d, int nRows) {
    int gw = blockIdx.x * 8 + (threadIdx.x >> 5);
    int lane = threadIdx.x & 31;
    if (gw >= nRows * HH) return;
    int n = gw / HH, h = gw % HH;
    const float* xrow = Hw + (size_t)n * DW;
    const float* vrow = v + h * DW;
    float acc = 0.0f;
    for (int c = lane; c < DW; c += 32) acc += xrow[c] * vrow[c];
    acc = wave_sum(acc);
    if (lane == 0) a_d[gw] = acc;
}

// alpha = leaky_relu(a_s[src]+a_d[dst]); segment max via monotone-uint atomicMax
__global__ void k_alpha(const long long* __restrict__ s2w, const float* __restrict__ a_s,
                        const float* __restrict__ a_d, float* __restrict__ alpha,
                        unsigned int* __restrict__ menc, int E) {
    int idx = blockIdx.x * blockDim.x + threadIdx.x;
    if (idx >= E * HH) return;
    int e = idx / HH, h = idx % HH;
    int s = (int)s2w[e];
    int d = (int)s2w[E + e];
    float a = a_s[s * HH + h] + a_d[d * HH + h];
    a = a > 0.0f ? a : 0.2f * a;
    alpha[idx] = a;
    atomicMax(&menc[d * HH + h], enc_f32(a));
}

// ex = exp(alpha - m[dst]) in-place; den[dst] += ex
__global__ void k_ex(const long long* __restrict__ s2w, float* __restrict__ alpha,
                     const unsigned int* __restrict__ menc, float* __restrict__ den, int E) {
    int idx = blockIdx.x * blockDim.x + threadIdx.x;
    if (idx >= E * HH) return;
    int e = idx / HH, h = idx % HH;
    int d = (int)s2w[E + e];
    float x = __expf(alpha[idx] - dec_f32(menc[d * HH + h]));
    alpha[idx] = x;
    atomicAdd(&den[d * HH + h], x);
}

// agg[dst, :] += xs[src, :] * w[h]   (block per edge, f32 atomics in L2)
__global__ __launch_bounds__(256) void k_scatter(const long long* __restrict__ s2w,
                                                 const float* __restrict__ ex,
                                                 const float* __restrict__ den,
                                                 const float* __restrict__ xs,
                                                 float* __restrict__ agg, int E) {
    int e = blockIdx.x;
    int s = (int)s2w[e];
    int d = (int)s2w[E + e];
    __shared__ float w[HH];
    if (threadIdx.x < HH)
        w[threadIdx.x] = ex[e * HH + threadIdx.x] / den[d * HH + threadIdx.x];
    __syncthreads();
    const float* xrow = xs + (size_t)s * HCP;
    float* arow = agg + (size_t)d * HC;
    for (int i = threadIdx.x; i < HC; i += 256)
        atomicAdd(&arow[i], xrow[i] * w[i / CC]);
}

// Uw = elu(agg + bias_gat) -> bf16 with K-pad columns forced to zero
__global__ void k_elu(const float* __restrict__ agg, const float* __restrict__ bias,
                      unsigned short* __restrict__ Uw, int nRows) {
    long long idx = blockIdx.x * (long long)blockDim.x + threadIdx.x;
    if (idx >= (long long)nRows * HCP) return;
    int i = (int)(idx % HCP);
    long long n = idx / HCP;
    float u = 0.0f;
    if (i < HC) {
        u = agg[n * HC + i] + bias[i];
        u = u > 0.0f ? u : expm1f(u);
    }
    Uw[idx] = f2bf(u);
}

// ---------------------------------------------------------------------------
// Host orchestration
// ---------------------------------------------------------------------------
static inline char* carve(char*& p, size_t bytes) {
    char* r = p;
    p += (bytes + 255) & ~(size_t)255;
    return r;
}

extern "C" void kernel_launch(void* const* d_in, const int* in_sizes, int n_in,
                              void* d_out, int out_size, void* d_ws, size_t ws_size,
                              hipStream_t stream) {
    const float*     Hw       = (const float*)d_in[0];
    const float*     Hs       = (const float*)d_in[1];
    const long long* s2w      = (const long long*)d_in[2];
    const float*     Wsrc     = (const float*)d_in[3];
    const float*     Wdst     = (const float*)d_in[4];
    const float*     att_src  = (const float*)d_in[5];
    const float*     att_dst  = (const float*)d_in[6];
    const float*     bias_gat = (const float*)d_in[7];
    const float*     W1       = (const float*)d_in[8];
    const float*     b1       = (const float*)d_in[9];
    const float*     W2       = (const float*)d_in[10];
    const float*     b2       = (const float*)d_in[11];
    float*           out      = (float*)d_out;

    const int E  = in_sizes[2] / 2;
    const int nw = NW_, ns = NS_;

    // workspace carve
    char* p = (char*)d_ws;
    float*          xs      = (float*)         carve(p, (size_t)ns * HCP * 4);
    unsigned short* Hs_bf   = (unsigned short*)carve(p, (size_t)ns * DS * 2);
    unsigned short* Wsrc_bf = (unsigned short*)carve(p, (size_t)HCP * DS * 2);
    unsigned short* Wf_bf   = (unsigned short*)carve(p, (size_t)NP * HCP * 2);
    float*          bfv     = (float*)         carve(p, (size_t)NP * 4);
    float*          vdst    = (float*)         carve(p, (size_t)HH * DW * 4);
    float*          a_s     = (float*)         carve(p, (size_t)ns * HH * 4);
    float*          a_d     = (float*)         carve(p, (size_t)nw * HH * 4);
    float*          alpha   = (float*)         carve(p, (size_t)E * HH * 4);
    unsigned int*   menc    = (unsigned int*)  carve(p, (size_t)nw * HH * 4);
    float*          den     = (float*)         carve(p, (size_t)nw * HH * 4);
    float*          agg     = (float*)         carve(p, (size_t)nw * HC * 4);
    unsigned short* Uw_bf   = (unsigned short*)carve(p, (size_t)nw * HCP * 2);

    // --- zero accumulators (menc=0 encodes "below -inf") ---
    {
        long long n = (long long)nw * HC;
        k_zero_f32<<<(unsigned)((n + 255) / 256), 256, 0, stream>>>(agg, n);
        n = (long long)nw * HH;
        k_zero_f32<<<(unsigned)((n + 255) / 256), 256, 0, stream>>>((float*)menc, n);
        k_zero_f32<<<(unsigned)((n + 255) / 256), 256, 0, stream>>>(den, n);
    }

    // --- precompute / conversions ---
    {
        long long t = (long long)ns * DS;
        k_cvt_pad<<<(unsigned)((t + 255) / 256), 256, 0, stream>>>(Hs, Hs_bf, ns, DS, t);
        t = (long long)HCP * DS;
        k_cvt_pad<<<(unsigned)((t + 255) / 256), 256, 0, stream>>>(Wsrc, Wsrc_bf, HC, DS, t);
    }
    k_vdst<<<(HH * DW + 255) / 256, 256, 0, stream>>>(Wdst, att_dst, vdst);
    k_foldW<<<(NP * HCP + 255) / 256, 256, 0, stream>>>(W2, W1, Wf_bf);
    k_foldb<<<(NP + 255) / 256, 256, 0, stream>>>(W2, b1, b2, bfv);

    // --- xs = Hs @ Wsrc^T  (bf16 WMMA, f32 accum) ---
    {
        dim3 grid((HCP + 63) / 64, (ns + 63) / 64);
        k_wmma_gemm<<<grid, 128, 0, stream>>>(Hs_bf, DS, ns, Wsrc_bf, DS, HCP,
                                              xs, HCP, HCP, DS, nullptr, nullptr, 0);
    }

    // --- attention scalars ---
    k_as<<<(ns * HH + 7) / 8, 256, 0, stream>>>(xs, att_src, a_s, ns);
    k_ad<<<(nw * HH + 7) / 8, 256, 0, stream>>>(Hw, vdst, a_d, nw);

    // --- edge softmax ---
    k_alpha<<<(E * HH + 255) / 256, 256, 0, stream>>>(s2w, a_s, a_d, alpha, menc, E);
    k_ex<<<(E * HH + 255) / 256, 256, 0, stream>>>(s2w, alpha, menc, den, E);

    // --- message scatter ---
    k_scatter<<<E, 256, 0, stream>>>(s2w, alpha, den, xs, agg, E);

    // --- elu + bf16 convert ---
    {
        long long t = (long long)nw * HCP;
        k_elu<<<(unsigned)((t + 255) / 256), 256, 0, stream>>>(agg, bias_gat, Uw_bf, nw);
    }

    // --- out = Uw @ (W2@W1)^T + (W2@b1+b2) + Hw  (bf16 WMMA, fused epilogue) ---
    {
        dim3 grid((NP + 63) / 64, (nw + 63) / 64);
        k_wmma_gemm<<<grid, 128, 0, stream>>>(Uw_bf, HCP, nw, Wf_bf, HCP, NP,
                                              out, DW, DW, HCP, bfv, Hw, DW);
    }
}
